// WorldModel_69715909149380
// MI455X (gfx1250) — matrix operand order repaired and checked by reading, not compiled, not run on hardware
//
#include <hip/hip_runtime.h>

// ---------------------------------------------------------------------------
// RSSM world-model scan for gfx1250 (MI455X), bf16 WMMA, persistent kernel.
// ---------------------------------------------------------------------------

typedef unsigned short u16;
typedef unsigned int   u32;
typedef __attribute__((ext_vector_type(16))) __bf16 bf16x16;
typedef __attribute__((ext_vector_type(8)))  float  f32x8;

// Problem dims
constexpr int BB = 64, TT = 256, EE = 1536, AAD = 6, SS = 32, DD = 512, HH = 512;
constexpr int NWG = 16;          // persistent workgroups (1 per WGP)
constexpr int SLC = DD / NWG;    // 32 columns of H/D owned per WG

// ---- workspace layout (bytes) ----
constexpr size_t OFF_BAR     = 0;          // 1024 ints (768 used)
constexpr size_t OFF_XG      = 4096;       // bf16 [64][512]
constexpr size_t OFF_DG      = 69632;      // bf16 [64][512]
constexpr size_t OFF_HG      = 135168;     // bf16 [64][512]
constexpr size_t OFF_XOG     = 200704;     // bf16 [64][512]
constexpr size_t OFF_SW_INP  = 266240;     // swizzled W_inp (stoch part) 32x512
constexpr size_t OFF_SW_GRU  = 299008;     // swizzled W_gru 1024x1536
constexpr size_t OFF_SW_IMG  = 3444736;    // swizzled W_img 512x512
constexpr size_t OFF_SW_OBSD = 3969024;    // swizzled W_obs rows 0..511
constexpr size_t OFF_SW_OBSE = 4493312;    // swizzled W_obs rows 512..2047
constexpr size_t OFF_SW_IMS  = 6066176;    // swizzled W_ims 512x64
constexpr size_t OFF_SW_OBST = 6131712;    // swizzled W_obsstat 512x64
constexpr size_t OFF_EPROJ   = 6197248;    // f32 [T][B][H] = 32 MB

// ---- d_out layout (floats, tuple concat order) ----
constexpr size_t O_POST_STOCH = 0;
constexpr size_t O_DET        = 524288;
constexpr size_t O_POST_MEAN  = 8912896;
constexpr size_t O_POST_STD   = 9437184;
constexpr size_t O_PRI_STOCH  = 9961472;
constexpr size_t O_PRI_MEAN   = 10485760;
constexpr size_t O_PRI_STD    = 11010048;

// LDS bytes for scan kernel
constexpr unsigned SMEM_BYTES = 276480;

struct Q2     { uint4 a, b; };
struct U32x8  { u32 u[8]; };

static __device__ inline u16 f2b(float f) {           // f32 -> bf16 bits (RNE)
  u32 u = __float_as_uint(f);
  u += 0x7fffu + ((u >> 16) & 1u);
  return (u16)(u >> 16);
}

// Packed pair conversion: lo -> bits[15:0], hi -> bits[31:16].
static __device__ inline u32 f2b_pk(float lo, float hi) {
#if __has_builtin(__builtin_amdgcn_cvt_pk_bf16_f32)
  auto pk = __builtin_amdgcn_cvt_pk_bf16_f32(lo, hi);
  u32 r;
  __builtin_memcpy(&r, &pk, sizeof(r));
  return r;
#else
  return (u32)f2b(lo) | ((u32)f2b(hi) << 16);
#endif
}

static __device__ inline f32x8 wmma_bf(bf16x16 a, bf16x16 b, f32x8 c) {
  return __builtin_amdgcn_wmma_f32_16x16x32_bf16(false, a, false, b, (short)0, c,
                                                 false, false);
}

// Load a B/A fragment stored in swizzled "fragment order": 16 contiguous halfs
// per lane per 16x16x32 tile (lane*16 offset).
static __device__ inline bf16x16 frag_from(const u16* p16) {
  Q2 q;
  q.a = *(const uint4*)(p16);
  q.b = *(const uint4*)(p16 + 8);
  return __builtin_bit_cast(bf16x16, q);
}

// A-fragment (16x32 bf16) from a row-major [rows][ldk] bf16 buffer.
// Lane L: row m0 + (L&15); K-chunks [k0+8*(L>>4), +8) and [k0+16+8*(L>>4), +8).
static __device__ inline bf16x16 afrag(const u16* base, int ldk, int m0, int k0,
                                       int lane) {
  const u16* r = base + (size_t)(m0 + (lane & 15)) * ldk + k0 + 8 * (lane >> 4);
  Q2 q;
  q.a = *(const uint4*)(r);
  q.b = *(const uint4*)(r + 16);
  return __builtin_bit_cast(bf16x16, q);
}

static __device__ inline float sigm(float x) { return 1.f / (1.f + __expf(-x)); }
static __device__ inline float softplus_f(float x) {
  return (x > 20.f) ? x : __logf(1.f + __expf(x));
}
static __device__ inline float elu_f(float x) {
  return x > 0.f ? x : (__expf(x) - 1.f);
}

// Device-scope barrier across the 16 persistent workgroups.
static __device__ inline void gbar(int* bar, int slot, int nwg) {
  __threadfence();
  __syncthreads();
  if (threadIdx.x == 0) {
    __hip_atomic_fetch_add(bar + slot, 1, __ATOMIC_ACQ_REL,
                           __HIP_MEMORY_SCOPE_AGENT);
    while (__hip_atomic_load(bar + slot, __ATOMIC_ACQUIRE,
                             __HIP_MEMORY_SCOPE_AGENT) < nwg)
      __builtin_amdgcn_s_sleep(1);
  }
  __syncthreads();
  __threadfence();
}

// ---------------------------------------------------------------------------
// Kernel 0: zero barrier slots + deter_glob
// ---------------------------------------------------------------------------
__global__ void k_init(int* __restrict__ bar, u32* __restrict__ dg) {
  int i = blockIdx.x * 256 + threadIdx.x;
  if (i < 1024) bar[i] = 0;
  if (i < 16384) dg[i] = 0u;  // 64*512 bf16 = 16384 dwords
}

// ---------------------------------------------------------------------------
// Kernel 1: swizzle a row-major f32 weight [K][N] into bf16 fragment order:
//   dst layout [NT][KT][512 halfs]; element e of lane L of tile (nt,kt) is
//   W[kt*32 + 8*(L>>4) + (e&7) + 16*(e>>3)][nt*16 + (L&15)]      (B-matrix 32x16 layout)
// ---------------------------------------------------------------------------
__global__ void k_swz(const float* __restrict__ W, u16* __restrict__ dst,
                      int K, int N) {
  const int KT = K >> 5;
  const size_t total = (size_t)K * N;
  size_t i = (size_t)blockIdx.x * 256 + threadIdx.x;
  if (i >= total) return;
  int tile = (int)(i >> 9);
  int within = (int)(i & 511);
  int lane = within >> 4, e = within & 15;
  int nt = tile / KT, kt = tile - nt * KT;
  int k = kt * 32 + ((lane >> 4) << 3) + (e & 7) + ((e >> 3) << 4);
  int n = nt * 16 + (lane & 15);
  dst[i] = f2b(W[(size_t)k * N + n]);
}

// ---------------------------------------------------------------------------
// Kernel 2: eproj[t][b][:] = embed[b][t][:] @ W_obs_e + b_obs   (f32 out)
// grid (T, H/128), block 256 (8 waves). Block: 64 rows (one t) x 128 cols;
// each wave owns 1 M-tile x 4 N-tiles so one converted A-fragment feeds
// 4 WMMAs (keeps the f32->bf16 conversion off the critical VALU path).
// ---------------------------------------------------------------------------
__global__ __launch_bounds__(256) void k_eproj(const float* __restrict__ embed,
                                               const u16* __restrict__ wB,
                                               const float* __restrict__ bias,
                                               float* __restrict__ eproj) {
  const int t = blockIdx.x, by = blockIdx.y;
  const int lane = threadIdx.x & 31, w = threadIdx.x >> 5;
  const int mh = w & 3, hh = w >> 2;
  const int ln = lane & 15, lh = lane >> 4;

  f32x8 acc[4];
#pragma unroll
  for (int j = 0; j < 4; ++j) {
    float bv = bias[by * 128 + (4 * hh + j) * 16 + ln];
#pragma unroll
    for (int r = 0; r < 8; ++r) acc[j][r] = bv;
  }

  const int b_row = 16 * mh + ln;
  const float* arow = embed + ((size_t)b_row * TT + t) * EE;

  for (int kt = 0; kt < EE / 32; ++kt) {
    // A fragment with on-the-fly packed f32 -> bf16 conversion
    const float* pa = arow + kt * 32 + 8 * lh;
    float4 f0 = *(const float4*)(pa);
    float4 f1 = *(const float4*)(pa + 4);
    float4 f2 = *(const float4*)(pa + 16);
    float4 f3 = *(const float4*)(pa + 20);
    U32x8 ua;
    ua.u[0] = f2b_pk(f0.x, f0.y);
    ua.u[1] = f2b_pk(f0.z, f0.w);
    ua.u[2] = f2b_pk(f1.x, f1.y);
    ua.u[3] = f2b_pk(f1.z, f1.w);
    ua.u[4] = f2b_pk(f2.x, f2.y);
    ua.u[5] = f2b_pk(f2.z, f2.w);
    ua.u[6] = f2b_pk(f3.x, f3.y);
    ua.u[7] = f2b_pk(f3.z, f3.w);
    bf16x16 af = __builtin_bit_cast(bf16x16, ua);
#pragma unroll
    for (int j = 0; j < 4; ++j) {
      int nt = 8 * by + 4 * hh + j;
      const u16* pb = wB + (((size_t)nt * (EE / 32) + kt) << 9) + lane * 16;
      acc[j] = wmma_bf(af, frag_from(pb), acc[j]);
    }
  }
#pragma unroll
  for (int j = 0; j < 4; ++j) {
    int col = by * 128 + (4 * hh + j) * 16 + ln;
#pragma unroll
    for (int r = 0; r < 8; ++r) {
      int m = 16 * mh + 8 * lh + r;
      eproj[((size_t)t * BB + m) * HH + col] = acc[j][r];
    }
  }
}

// ---------------------------------------------------------------------------
// Kernel 3: persistent sequential scan (16 WGs, 256 thr = 8 waves each)
// ---------------------------------------------------------------------------
struct ScanP {
  const float *action, *noise_p, *noise_o;
  const float *W_inp, *b_inp, *b_gru, *b_img, *b_ims, *b_obsstat;
  const float* eproj;
  const u16 *sw_inp, *sw_gru, *sw_img, *sw_obsd, *sw_ims, *sw_obst;
  u16 *xg, *dg, *hg, *xog;
  int* bar;
  float* out;
};

__global__ __launch_bounds__(256) void k_scan(ScanP p) {
  extern __shared__ char smem[];
  u16* ls_gru  = (u16*)smem;            // [6][32][512] halfs = 196608 B
  u16* ls_img  = ls_gru + 98304;        // [2][16][512]       =  32768 B
  u16* ls_obsd = ls_img + 16384;        // [2][16][512]       =  32768 B
  u16* ls_inp  = ls_obsd + 16384;       // [2][1][512]        =   2048 B
  u16* ls_st   = ls_inp + 1024;         // stoch bf16 [64][32]=   4096 B
  float* ls_dt = (float*)(ls_st + 2048);// deter f32  [64][32]=   8192 B

  const int tid  = threadIdx.x;
  const int lane = tid & 31;
  const int w    = tid >> 5;
  const int mh   = w & 3;     // M-tile (rows 16*mh..)
  const int hh   = w >> 2;    // 0/1: column half of this WG's 32-col slice
  const int ln   = lane & 15;
  const int lh   = lane >> 4;
  const int wg   = blockIdx.x;
  const int colbase = wg * SLC;

  // ---- stage weight slices to LDS (fragment order, contiguous memcpy) ----
  for (int g = 0; g < 3; ++g)
    for (int i = 0; i < 2; ++i) {
      const u16* src = p.sw_gru + (size_t)(32 * g + 2 * wg + i) * 16384;
      u16* dst = ls_gru + (2 * g + i) * 16384;
      for (int x = tid * 8; x < 16384; x += 2048)
        *(uint4*)(dst + x) = *(const uint4*)(src + x);
    }
  for (int i = 0; i < 2; ++i) {
    const u16* s1 = p.sw_img + (size_t)(2 * wg + i) * 8192;
    const u16* s2 = p.sw_obsd + (size_t)(2 * wg + i) * 8192;
    for (int x = tid * 8; x < 8192; x += 2048) {
      *(uint4*)(ls_img + i * 8192 + x)  = *(const uint4*)(s1 + x);
      *(uint4*)(ls_obsd + i * 8192 + x) = *(const uint4*)(s2 + x);
    }
    const u16* s3 = p.sw_inp + (size_t)(2 * wg + i) * 512;
    for (int x = tid * 8; x < 512; x += 2048)
      *(uint4*)(ls_inp + i * 512 + x) = *(const uint4*)(s3 + x);
  }
  for (int x = tid; x < 2048; x += 256) { ls_st[x] = 0; ls_dt[x] = 0.f; }
  __syncthreads();

  for (int t = 0; t < TT; ++t) {
    // ---------- (1) x slice: elu(stoch @ W_inp_s + act_proj + b) ----------
    {
      const int col = colbase + 16 * hh + ln;
      float wk[6];
#pragma unroll
      for (int a = 0; a < 6; ++a) wk[a] = p.W_inp[(size_t)(SS + a) * HH + col];
      const float binp = p.b_inp[col];
      f32x8 c;
#pragma unroll
      for (int r = 0; r < 8; ++r) {
        int m = 16 * mh + 8 * lh + r;
        float s = binp;
#pragma unroll
        for (int a = 0; a < 6; ++a)
          s += p.action[((size_t)m * TT + t) * AAD + a] * wk[a];
        c[r] = s;
      }
      bf16x16 af = afrag(ls_st, SS, 16 * mh, 0, lane);
      bf16x16 bf = frag_from(ls_inp + (hh << 9) + lane * 16);
      c = wmma_bf(af, bf, c);
#pragma unroll
      for (int r = 0; r < 8; ++r) {
        int m = 16 * mh + 8 * lh + r;
        p.xg[(size_t)m * HH + col] = f2b(elu_f(c[r]));
      }
    }
    gbar(p.bar, 3 * t + 0, NWG);

    // ---------- (2) GRU: parts = [x,deter] @ W_gru + b ----------
    {
      const int nl = 16 * hh + ln;       // local col in slice
      const int col = colbase + nl;      // col in 0..511
      f32x8 acc[3];
#pragma unroll
      for (int g = 0; g < 3; ++g) {
        float bg = p.b_gru[(size_t)g * DD + col];
#pragma unroll
        for (int r = 0; r < 8; ++r) acc[g][r] = bg;
      }
      for (int kk = 0; kk < 32; ++kk) {
        bf16x16 af = (kk < 16)
                         ? afrag(p.xg, HH, 16 * mh, kk * 32, lane)
                         : afrag(p.dg, DD, 16 * mh, (kk - 16) * 32, lane);
#pragma unroll
        for (int g = 0; g < 3; ++g) {
          const u16* pb =
              ls_gru + ((((2 * g + hh) * 32 + kk)) << 9) + lane * 16;
          acc[g] = wmma_bf(af, frag_from(pb), acc[g]);
        }
      }
#pragma unroll
      for (int r = 0; r < 8; ++r) {
        int m = 16 * mh + 8 * lh + r;
        float reset = sigm(acc[0][r]);
        float cand  = tanhf(reset * acc[1][r]);
        float upd   = sigm(acc[2][r] - 1.f);
        float dprev = ls_dt[m * SLC + nl];
        float dn = upd * cand + (1.f - upd) * dprev;
        ls_dt[m * SLC + nl] = dn;
        p.dg[(size_t)m * DD + col] = f2b(dn);
        p.out[O_DET + ((size_t)m * TT + t) * DD + col] = dn;   // deter [B,T,D]
      }
    }
    gbar(p.bar, 3 * t + 1, NWG);

    // ---------- (3) img + obs: h = elu(dn@W_img+b); xo = elu(dn@W_obs_d+eproj)
    {
      const int col = colbase + 16 * hh + ln;
      const float bi = p.b_img[col];
      f32x8 ch, co;
#pragma unroll
      for (int r = 0; r < 8; ++r) {
        int m = 16 * mh + 8 * lh + r;
        ch[r] = bi;
        co[r] = p.eproj[((size_t)t * BB + m) * HH + col];  // b_obs included
      }
      for (int kk = 0; kk < 16; ++kk) {
        bf16x16 af = afrag(p.dg, DD, 16 * mh, kk * 32, lane);
        bf16x16 b1 = frag_from(ls_img + ((hh * 16 + kk) << 9) + lane * 16);
        bf16x16 b2 = frag_from(ls_obsd + ((hh * 16 + kk) << 9) + lane * 16);
        ch = wmma_bf(af, b1, ch);
        co = wmma_bf(af, b2, co);
      }
#pragma unroll
      for (int r = 0; r < 8; ++r) {
        int m = 16 * mh + 8 * lh + r;
        p.hg[(size_t)m * HH + col]  = f2b(elu_f(ch[r]));
        p.xog[(size_t)m * HH + col] = f2b(elu_f(co[r]));
      }
    }
    gbar(p.bar, 3 * t + 2, NWG);

    // ---------- (4) posterior stats: all WGs redundantly (carried stoch) ----
    {
      const int s = 16 * hh + ln;
      f32x8 cm, cs;
      const float bm = p.b_obsstat[s], bs = p.b_obsstat[SS + s];
#pragma unroll
      for (int r = 0; r < 8; ++r) { cm[r] = bm; cs[r] = bs; }
      for (int kk = 0; kk < 16; ++kk) {
        bf16x16 af = afrag(p.xog, HH, 16 * mh, kk * 32, lane);
        const u16* pm_ = p.sw_obst + ((hh * 16 + kk) << 9) + lane * 16;
        const u16* ps_ = p.sw_obst + (((hh + 2) * 16 + kk) << 9) + lane * 16;
        if (kk < 15) __builtin_prefetch(pm_ + 512, 0, 1);
        cm = wmma_bf(af, frag_from(pm_), cm);
        cs = wmma_bf(af, frag_from(ps_), cs);
      }
#pragma unroll
      for (int r = 0; r < 8; ++r) {
        int m = 16 * mh + 8 * lh + r;
        float mean = cm[r];
        float sd = softplus_f(cs[r]) + 0.1f;
        float nz = p.noise_o[((size_t)t * BB + m) * SS + s];
        float st = mean + sd * nz;
        ls_st[m * SS + s] = f2b(st);          // carried posterior sample
        if (wg == 0) {
          size_t o = ((size_t)m * TT + t) * SS + s;
          p.out[O_POST_MEAN + o]  = mean;
          p.out[O_POST_STD + o]   = sd;
          p.out[O_POST_STOCH + o] = st;
        }
      }
    }

    // ---------- (5) prior stats: WG1 only (output-only path) ----------
    if (wg == 1) {
      const int s = 16 * hh + ln;
      f32x8 cm, cs;
      const float bm = p.b_ims[s], bs = p.b_ims[SS + s];
#pragma unroll
      for (int r = 0; r < 8; ++r) { cm[r] = bm; cs[r] = bs; }
      for (int kk = 0; kk < 16; ++kk) {
        bf16x16 af = afrag(p.hg, HH, 16 * mh, kk * 32, lane);
        const u16* pm_ = p.sw_ims + ((hh * 16 + kk) << 9) + lane * 16;
        const u16* ps_ = p.sw_ims + (((hh + 2) * 16 + kk) << 9) + lane * 16;
        cm = wmma_bf(af, frag_from(pm_), cm);
        cs = wmma_bf(af, frag_from(ps_), cs);
      }
#pragma unroll
      for (int r = 0; r < 8; ++r) {
        int m = 16 * mh + 8 * lh + r;
        float mean = cm[r];
        float sd = softplus_f(cs[r]) + 0.1f;
        float nz = p.noise_p[((size_t)t * BB + m) * SS + s];
        size_t o = ((size_t)m * TT + t) * SS + s;
        p.out[O_PRI_MEAN + o]  = mean;
        p.out[O_PRI_STD + o]   = sd;
        p.out[O_PRI_STOCH + o] = mean + sd * nz;
      }
    }
    __syncthreads();  // protect ls_st before next step's x-GEMM
  }
}

// ---------------------------------------------------------------------------
extern "C" void kernel_launch(void* const* d_in, const int* in_sizes, int n_in,
                              void* d_out, int out_size, void* d_ws,
                              size_t ws_size, hipStream_t stream) {
  const float* embed     = (const float*)d_in[0];
  const float* action    = (const float*)d_in[1];
  const float* noise_p   = (const float*)d_in[2];
  const float* noise_o   = (const float*)d_in[3];
  const float* W_inp     = (const float*)d_in[4];
  const float* b_inp     = (const float*)d_in[5];
  const float* W_gru     = (const float*)d_in[6];
  const float* b_gru     = (const float*)d_in[7];
  const float* W_img     = (const float*)d_in[8];
  const float* b_img     = (const float*)d_in[9];
  const float* W_ims     = (const float*)d_in[10];
  const float* b_ims     = (const float*)d_in[11];
  const float* W_obs     = (const float*)d_in[12];
  const float* b_obs     = (const float*)d_in[13];
  const float* W_obsstat = (const float*)d_in[14];
  const float* b_obsstat = (const float*)d_in[15];

  unsigned char* ws = (unsigned char*)d_ws;
  int* bar      = (int*)(ws + OFF_BAR);
  u16* xg       = (u16*)(ws + OFF_XG);
  u16* dg       = (u16*)(ws + OFF_DG);
  u16* hg       = (u16*)(ws + OFF_HG);
  u16* xog      = (u16*)(ws + OFF_XOG);
  u16* sw_inp   = (u16*)(ws + OFF_SW_INP);
  u16* sw_gru   = (u16*)(ws + OFF_SW_GRU);
  u16* sw_img   = (u16*)(ws + OFF_SW_IMG);
  u16* sw_obsd  = (u16*)(ws + OFF_SW_OBSD);
  u16* sw_obse  = (u16*)(ws + OFF_SW_OBSE);
  u16* sw_ims   = (u16*)(ws + OFF_SW_IMS);
  u16* sw_obst  = (u16*)(ws + OFF_SW_OBST);
  float* eproj  = (float*)(ws + OFF_EPROJ);

  // 0) zero barrier slots + deter_glob
  k_init<<<dim3(64), 256, 0, stream>>>(bar, (u32*)dg);

  // 1) swizzle all weights into bf16 fragment order
  auto blocks = [](size_t n) { return (unsigned)((n + 255) / 256); };
  k_swz<<<blocks((size_t)32 * 512), 256, 0, stream>>>(W_inp, sw_inp, 32, 512);
  k_swz<<<blocks((size_t)1024 * 1536), 256, 0, stream>>>(W_gru, sw_gru, 1024, 1536);
  k_swz<<<blocks((size_t)512 * 512), 256, 0, stream>>>(W_img, sw_img, 512, 512);
  k_swz<<<blocks((size_t)512 * 512), 256, 0, stream>>>(W_obs, sw_obsd, 512, 512);
  k_swz<<<blocks((size_t)1536 * 512), 256, 0, stream>>>(W_obs + (size_t)512 * 512,
                                                        sw_obse, 1536, 512);
  k_swz<<<blocks((size_t)512 * 64), 256, 0, stream>>>(W_ims, sw_ims, 512, 64);
  k_swz<<<blocks((size_t)512 * 64), 256, 0, stream>>>(W_obsstat, sw_obst, 512, 64);

  // 2) embed projection (time-major), bias folded in
  k_eproj<<<dim3(TT, HH / 128), 256, 0, stream>>>(embed, sw_obse, b_obs, eproj);

  // 3) persistent sequential scan
  (void)hipFuncSetAttribute(reinterpret_cast<const void*>(k_scan),
                            hipFuncAttributeMaxDynamicSharedMemorySize,
                            (int)SMEM_BYTES);
  ScanP p;
  p.action = action; p.noise_p = noise_p; p.noise_o = noise_o;
  p.W_inp = W_inp; p.b_inp = b_inp; p.b_gru = b_gru; p.b_img = b_img;
  p.b_ims = b_ims; p.b_obsstat = b_obsstat; p.eproj = eproj;
  p.sw_inp = sw_inp; p.sw_gru = sw_gru; p.sw_img = sw_img;
  p.sw_obsd = sw_obsd; p.sw_ims = sw_ims; p.sw_obst = sw_obst;
  p.xg = xg; p.dg = dg; p.hg = hg; p.xog = xog;
  p.bar = bar; p.out = (float*)d_out;
  k_scan<<<dim3(NWG), 256, SMEM_BYTES, stream>>>(p);
}
